// VRWKV_SpatialMix_5050881540339
// MI455X (gfx1250) — compile-verified, matrix operand order checked by reading, not run on hardware
//
#include <hip/hip_runtime.h>
#include <hip/hip_bf16.h>
#include <math.h>

// Problem constants (B, D, H, W, C) = (8, 16, 16, 16, 256)
#define Bsz 8
#define Tt  4096          // D*H*W
#define Cc  256
#define Mrows (Bsz * Tt)  // 32768
#define MTILE 128
#define PAD 8
#define ASTRIDE (Cc + PAD)                 // 264 ushorts per row (conflict padding)
#define SW_BYTES ((size_t)Cc * ASTRIDE * 2)      // 135168
#define SA_BYTES ((size_t)MTILE * ASTRIDE * 2)   // 67584
#define SMEM_BYTES (SW_BYTES + SA_BYTES)         // 202752 (< 320KB WGP LDS)

// Chunked-scan parameters for the WKV recurrence
#define NCH 16
#define CHL (Tt / NCH)    // 256 steps per chunk
#define NBC (Bsz * Cc)    // 2048 independent (b,c) scans

typedef __attribute__((ext_vector_type(16))) __bf16 v16bf;
typedef __attribute__((ext_vector_type(8)))  float  v8f;

union Frag { unsigned int u[8]; v16bf v; };

__device__ inline unsigned short f2bf(float f) {
  unsigned int u = __float_as_uint(f);
  u += 0x7FFFu + ((u >> 16) & 1u);   // round-to-nearest-even
  return (unsigned short)(u >> 16);
}

// WMMA GEMM body: out[m0+128 rows][256] = A(128x256 bf16, LDS) * W^T.
// A-fragments (64 VGPRs) stay resident; two N-tiles are processed per
// iteration with independent accumulators so one tile's WMMAs overlap the
// other tile's ds_load_b128 fragment fetches (hides s_wait_dscnt stalls).
__device__ inline void gemm_tile_compute(const unsigned short* sA,
                                         const unsigned short* sW,
                                         float* __restrict__ outp, int m0) {
  const int tid   = threadIdx.x;
  const int wave  = tid >> 5;
  const int lane  = tid & 31;
  const int lrow  = lane & 15;
  const int khalf = lane >> 4;
  const int row0  = wave * 16;
  const int kbase = khalf * 8;

  // Load all 8 A fragments once (rows resident across the whole N sweep).
  const unsigned short* arow = sA + (row0 + lrow) * ASTRIDE;
  Frag a[8];
#pragma unroll
  for (int kb = 0; kb < 8; ++kb) {
    const unsigned short* ap = arow + kb * 32 + kbase;
#pragma unroll
    for (int p = 0; p < 4; ++p) {
      a[kb].u[p]     = *(const unsigned int*)(ap + 2 * p);        // K pairs 0..7
      a[kb].u[p + 4] = *(const unsigned int*)(ap + 16 + 2 * p);   // K pairs 16..23
    }
  }

  for (int nt = 0; nt < 16; nt += 2) {
    const unsigned short* br0 = sW + (nt * 16 + lrow) * ASTRIDE + kbase;
    const unsigned short* br1 = br0 + 16 * ASTRIDE;
    v8f acc0 = {}, acc1 = {};
#pragma unroll
    for (int kb = 0; kb < 8; ++kb) {
      Frag b0, b1;
#pragma unroll
      for (int p = 0; p < 4; ++p) {
        b0.u[p]     = *(const unsigned int*)(br0 + kb * 32 + 2 * p);
        b0.u[p + 4] = *(const unsigned int*)(br0 + kb * 32 + 16 + 2 * p);
        b1.u[p]     = *(const unsigned int*)(br1 + kb * 32 + 2 * p);
        b1.u[p + 4] = *(const unsigned int*)(br1 + kb * 32 + 16 + 2 * p);
      }
      acc0 = __builtin_amdgcn_wmma_f32_16x16x32_bf16(
          false, a[kb].v, false, b0.v, (short)0, acc0, false, false);
      acc1 = __builtin_amdgcn_wmma_f32_16x16x32_bf16(
          false, a[kb].v, false, b1.v, (short)0, acc1, false, false);
    }
#pragma unroll
    for (int j = 0; j < 8; ++j) {
      const size_t orow = (size_t)(m0 + row0 + j + khalf * 8);
      outp[orow * Cc + nt * 16 + lrow]       = acc0[j];
      outp[orow * Cc + (nt + 1) * 16 + lrow] = acc1[j];
    }
  }
}

// Kernel 1: fused q_shift_3d + token-mix + three GEMMs (k/v/r via blockIdx.y)
__global__ void mix_gemm_kernel(const float* __restrict__ x,
                                const float* __restrict__ mix_k,
                                const float* __restrict__ mix_v,
                                const float* __restrict__ mix_r,
                                const float* __restrict__ W_key,
                                const float* __restrict__ W_value,
                                const float* __restrict__ W_recept,
                                float* __restrict__ kbuf,
                                float* __restrict__ vbuf,
                                float* __restrict__ rbuf) {
  extern __shared__ __align__(16) char smem[];
  unsigned short* sW = (unsigned short*)smem;
  unsigned short* sA = (unsigned short*)(smem + SW_BYTES);
  const int tid = threadIdx.x;
  const int which = blockIdx.y;
  const float* mix = (which == 0) ? mix_k : ((which == 1) ? mix_v : mix_r);
  const float* W   = (which == 0) ? W_key : ((which == 1) ? W_value : W_recept);
  float* outp      = (which == 0) ? kbuf  : ((which == 1) ? vbuf    : rbuf);
  const int m0 = blockIdx.x * MTILE;

  for (int i = tid; i < Cc * Cc; i += 256) {
    __builtin_prefetch(W + i + 4096, 0, 1);   // global_prefetch_b8
    sW[(i >> 8) * ASTRIDE + (i & 255)] = f2bf(W[i]);
  }
  // A tile: x*mix + shift(x)*(1-mix). gamma=0.25 -> exactly 4 groups of 64:
  //   c/64==0: W-1 | 1: W+1 | 2: H-1 | 3: H+1  (zero outside boundary)
  for (int i = tid; i < MTILE * Cc; i += 256) {
    const int r = i >> 8;
    const int c = i & 255;
    const int grow = m0 + r;
    const int b = grow >> 12;       // /T
    const int t = grow & 4095;
    const int w = t & 15;
    const int h = (t >> 4) & 15;
    const int g = c >> 6;
    int tsrc; bool valid;
    if (g == 0)      { tsrc = t - 1;  valid = (w > 0);  }
    else if (g == 1) { tsrc = t + 1;  valid = (w < 15); }
    else if (g == 2) { tsrc = t - 16; valid = (h > 0);  }
    else             { tsrc = t + 16; valid = (h < 15); }
    const float xc = x[(size_t)grow * Cc + c];
    const float xs = valid ? x[((size_t)(b << 12) + tsrc) * Cc + c] : 0.f;
    const float mm = mix[c];
    sA[r * ASTRIDE + c] = f2bf(xc * mm + xs * (1.f - mm));
  }
  __syncthreads();
  gemm_tile_compute(sA, sW, outp, m0);
}

// ---- Chunked WKV scan ------------------------------------------------------
// State (p,q,o) is the log-scaled form of (N,D): N = p*e^o, D = q*e^o, with
// the linear recurrence N' = e^w N + e^k v, D' = e^w D + e^k. A chunk of L
// steps therefore acts as state_out = e^{Lw} * state_in + local, which makes
// the scan associative -> 3 passes.

// Pass A: chunk-local transforms from the zero state. 32768 threads.
__global__ void wkv_local_kernel(const float* __restrict__ sd,
                                 const float* __restrict__ kbuf,
                                 const float* __restrict__ vbuf,
                                 float* __restrict__ pbuf,
                                 float* __restrict__ qbuf,
                                 float* __restrict__ obuf) {
  const int g  = blockIdx.x * blockDim.x + threadIdx.x;  // 0..NCH*NBC-1
  const int ch = g >> 11;          // chunk
  const int bc = g & (NBC - 1);
  const int b  = bc >> 8;
  const int c  = bc & 255;
  const float wv = sd[c] * (1.f / Tt);
  float p = 0.f, q = 0.f, o = -1e38f;
  size_t idx = ((size_t)b * Tt + (size_t)ch * CHL) * Cc + c;
  for (int t = 0; t < CHL; ++t, idx += Cc) {
    const float kt = kbuf[idx];
    const float vt = vbuf[idx];
    const float no2 = fmaxf(wv + o, kt);
    const float A2  = __expf(wv + o - no2);
    const float B2  = __expf(kt - no2);
    p = A2 * p + B2 * vt;
    q = A2 * q + B2;
    o = no2;
  }
  pbuf[g] = p; qbuf[g] = q; obuf[g] = o;
}

// Pass B: sequential combine over NCH=16 chunks per (b,c); rewrites the
// chunk buffers in place with each chunk's *initial* state. 2048 threads.
__global__ void wkv_combine_kernel(const float* __restrict__ sd,
                                   float* __restrict__ pbuf,
                                   float* __restrict__ qbuf,
                                   float* __restrict__ obuf) {
  const int g2 = blockIdx.x * blockDim.x + threadIdx.x;  // 0..NBC-1
  const int c  = g2 & 255;
  const float Lw = sd[c] * ((float)CHL / Tt);            // L * w
  float p = 0.f, q = 0.f, o = -1e38f;
  for (int j = 0; j < NCH; ++j) {
    const int idx = j * NBC + g2;
    const float pl = pbuf[idx], ql = qbuf[idx], ol = obuf[idx];
    pbuf[idx] = p; qbuf[idx] = q; obuf[idx] = o;         // initial state of chunk j
    const float os = o + Lw;
    const float on = fmaxf(os, ol);
    const float e1 = __expf(os - on);
    const float e2 = __expf(ol - on);
    p = e1 * p + e2 * pl;
    q = e1 * q + e2 * ql;
    o = on;
  }
}

// Pass C: replay each chunk from its correct initial state, emit y. 32768 thr.
__global__ void wkv_emit_kernel(const float* __restrict__ sd,
                                const float* __restrict__ sf,
                                const float* __restrict__ kbuf,
                                const float* __restrict__ vbuf,
                                const float* __restrict__ pbuf,
                                const float* __restrict__ qbuf,
                                const float* __restrict__ obuf,
                                float* __restrict__ ybuf) {
  const int g  = blockIdx.x * blockDim.x + threadIdx.x;
  const int ch = g >> 11;
  const int bc = g & (NBC - 1);
  const int b  = bc >> 8;
  const int c  = bc & 255;
  const float wv = sd[c] * (1.f / Tt);
  const float uu = sf[c] * (1.f / Tt);
  float p = pbuf[g], q = qbuf[g], o = obuf[g];
  size_t idx = ((size_t)b * Tt + (size_t)ch * CHL) * Cc + c;
  for (int t = 0; t < CHL; ++t, idx += Cc) {
    const float kt = kbuf[idx];
    const float vt = vbuf[idx];
    const float no = fmaxf(o, uu + kt);
    const float A  = __expf(o - no);
    const float Bc = __expf(uu + kt - no);
    ybuf[idx] = (A * p + Bc * vt) / (A * q + Bc);
    const float no2 = fmaxf(wv + o, kt);
    const float A2  = __expf(wv + o - no2);
    const float B2  = __expf(kt - no2);
    p = A2 * p + B2 * vt;
    q = A2 * q + B2;
    o = no2;
  }
}

// Kernel 3: fused LayerNorm + sigmoid(r) gate + output GEMM.
__global__ void ln_out_gemm_kernel(const float* __restrict__ ybuf,
                                   const float* __restrict__ rbuf,
                                   const float* __restrict__ gamma,
                                   const float* __restrict__ beta,
                                   const float* __restrict__ W_out,
                                   float* __restrict__ outp) {
  extern __shared__ __align__(16) char smem[];
  unsigned short* sW = (unsigned short*)smem;
  unsigned short* sA = (unsigned short*)(smem + SW_BYTES);
  const int tid  = threadIdx.x;
  const int wave = tid >> 5;
  const int lane = tid & 31;
  const int m0   = blockIdx.x * MTILE;

  for (int i = tid; i < Cc * Cc; i += 256) {
    __builtin_prefetch(W_out + i + 4096, 0, 1);
    sW[(i >> 8) * ASTRIDE + (i & 255)] = f2bf(W_out[i]);
  }

  // Each wave handles 16 rows; 32 lanes x 8 channels, wave32 shuffles for LN.
  for (int rr = 0; rr < 16; ++rr) {
    const int rrel = wave * 16 + rr;
    const size_t base = (size_t)(m0 + rrel) * Cc;
    float vals[8];
    float s = 0.f;
#pragma unroll
    for (int i = 0; i < 8; ++i) { vals[i] = ybuf[base + lane + 32 * i]; s += vals[i]; }
#pragma unroll
    for (int off = 16; off; off >>= 1) s += __shfl_xor(s, off, 32);
    const float mean = s * (1.f / 256.f);
    float s2 = 0.f;
#pragma unroll
    for (int i = 0; i < 8; ++i) { const float d = vals[i] - mean; s2 += d * d; }
#pragma unroll
    for (int off = 16; off; off >>= 1) s2 += __shfl_xor(s2, off, 32);
    const float rstd = rsqrtf(s2 * (1.f / 256.f) + 1e-5f);
#pragma unroll
    for (int i = 0; i < 8; ++i) {
      const int c = lane + 32 * i;
      const float ln = (vals[i] - mean) * rstd * gamma[c] + beta[c];
      const float sr = 1.f / (1.f + __expf(-rbuf[base + c]));
      sA[rrel * ASTRIDE + c] = f2bf(sr * ln);
    }
  }
  __syncthreads();
  gemm_tile_compute(sA, sW, outp, m0);
}

extern "C" void kernel_launch(void* const* d_in, const int* in_sizes, int n_in,
                              void* d_out, int out_size, void* d_ws, size_t ws_size,
                              hipStream_t stream) {
  (void)in_sizes; (void)n_in; (void)out_size; (void)ws_size;
  const float* x  = (const float*)d_in[0];
  const float* sd = (const float*)d_in[1];
  const float* sf = (const float*)d_in[2];
  const float* mk = (const float*)d_in[3];
  const float* mv = (const float*)d_in[4];
  const float* mr = (const float*)d_in[5];
  const float* Wk = (const float*)d_in[6];
  const float* Wv = (const float*)d_in[7];
  const float* Wr = (const float*)d_in[8];
  const float* Wo = (const float*)d_in[9];
  const float* lg = (const float*)d_in[10];
  const float* lb = (const float*)d_in[11];
  float* out = (float*)d_out;

  const size_t NBUF = (size_t)Mrows * Cc;        // 8M floats = 32MB each
  float* kbuf = (float*)d_ws;
  float* vbuf = kbuf + NBUF;
  float* rbuf = vbuf + NBUF;
  float* ybuf = rbuf + NBUF;
  float* pbuf = ybuf + NBUF;                     // NCH*NBC floats (128KB) each
  float* qbuf = pbuf + (size_t)NCH * NBC;
  float* obuf = qbuf + (size_t)NCH * NBC;

  mix_gemm_kernel<<<dim3(Mrows / MTILE, 3), 256, SMEM_BYTES, stream>>>(
      x, mk, mv, mr, Wk, Wv, Wr, kbuf, vbuf, rbuf);
  wkv_local_kernel<<<dim3((NCH * NBC) / 256), 256, 0, stream>>>(
      sd, kbuf, vbuf, pbuf, qbuf, obuf);
  wkv_combine_kernel<<<dim3(NBC / 256), 256, 0, stream>>>(sd, pbuf, qbuf, obuf);
  wkv_emit_kernel<<<dim3((NCH * NBC) / 256), 256, 0, stream>>>(
      sd, sf, kbuf, vbuf, pbuf, qbuf, obuf, ybuf);
  ln_out_gemm_kernel<<<dim3(Mrows / MTILE), 256, SMEM_BYTES, stream>>>(
      ybuf, rbuf, lg, lb, Wo, out);
}
